// CloneNet_2396591751946
// MI455X (gfx1250) — matrix operand order repaired and checked by reading, not compiled, not run on hardware
//
#include <hip/hip_runtime.h>
#include <hip/hip_bf16.h>

// MPNN (NNConv + GRU + Set2Set) for MI455X / gfx1250, wave32, WMMA f16->f32.
//
// Key design points:
//  * The per-edge 64x64 NNConv weight matrices (w_e, 1.31 GB fp32 if
//    materialized) NEVER touch HBM. Each 256-thread workgroup recomputes its
//    16-edge x 4096 tile of w_e with v_wmma_f32_16x16x32_f16 into a 128 KB LDS
//    tile (possible only with CDNA5's 320 KB/WGP LDS) and consumes it
//    immediately for the message matvec: ~5.2 GB of HBM traffic becomes
//    ~252 GFLOP of f16 WMMA recompute, negligible against the WMMA ceiling.
//  * All GEMM B-operands (weights) are pre-packed ONCE into f16 buffers in the
//    exact WMMA B-fragment layout, so the hot loops issue one aligned 32-byte
//    vector load per fragment instead of 16 strided f32 loads + 8 converts.

#define NNODE 20000
#define NEDGE 80000
#define NB    64
#define HF    64      // H
#define NFD   32      // n_features
#define NHID  128     // edge-MLP hidden
#define HH    4096    // H*H

typedef __attribute__((ext_vector_type(16))) _Float16 v16h;
typedef __attribute__((ext_vector_type(4)))  _Float16 v4h;
typedef __attribute__((ext_vector_type(8)))  float    v8f;

// ---- WMMA fragment helpers -------------------------------------------------
// A (16x32 f16): lane l -> row l&15; halves: K = kc + (l>>4)*16 + {0..15}
__device__ __forceinline__ v16h frag_a_f32(const float* A, int lda, int row0,
                                           int kc, int lane) {
  int r  = row0 + (lane & 15);
  int kb = kc + ((lane >> 4) << 4);
  const float* p = A + (long)r * lda + kb;
  v16h a;
#pragma unroll
  for (int i = 0; i < 8; ++i) {
    a[i]     = (_Float16)p[i];
    a[i + 8] = (_Float16)p[8 + i];
  }
  return a;
}

// Packed-B fragment: one aligned 32-byte load from the pre-packed f16 buffer.
__device__ __forceinline__ v16h frag_b_pk(const _Float16* __restrict__ P,
                                          int ntiles, int c, int nt, int lane) {
  return *(const v16h*)(P + (((size_t)(c * ntiles + nt)) * 32 + lane) * 16);
}

__device__ __forceinline__ v8f wmma16(v16h a, v16h b, v8f c) {
  return __builtin_amdgcn_wmma_f32_16x16x32_f16(false, a, false, b, (short)0, c,
                                                false, false);
}

__device__ __forceinline__ float sigf(float x) {
  return 1.0f / (1.0f + __expf(-x));
}

// ---- One-time weight pack: f32 row-major [K,N] -> f16 B-fragment layout ----
// dst[((c*(N/16)+nt)*32 + lane)*16 + h] = src[(c*32 + (lane>>4)*16 + h)*N
//                                             + nt*16 + (lane&15)]
__global__ void k_pack(const float* __restrict__ src, _Float16* __restrict__ dst,
                       int K, int N) {
  int idx = blockIdx.x * 256 + threadIdx.x;
  if (idx >= K * N) return;
  int ntiles = N >> 4;
  int hh   = idx & 15;
  int lane = (idx >> 4) & 31;
  int t    = idx >> 9;
  int nt   = t % ntiles;
  int c    = t / ntiles;
  int k = c * 32 + ((lane >> 4) << 4) + hh;
  int n = nt * 16 + (lane & 15);
  dst[idx] = (_Float16)src[(long)k * N + n];
}

// ---- out = relu(x @ lin0_w + b); h = out.  M=20000, K=32, N=64 -------------
__global__ void k_lin0(const float* __restrict__ x,
                       const _Float16* __restrict__ wp,
                       const float* __restrict__ bias, float* __restrict__ out,
                       float* __restrict__ h) {
  int lane = threadIdx.x;
  int mt = blockIdx.x >> 2, nt = blockIdx.x & 3;
  int row0 = mt * 16, col0 = nt * 16;
  int n = col0 + (lane & 15);
  float bv = bias[n];
  v8f c = {bv, bv, bv, bv, bv, bv, bv, bv};
  c = wmma16(frag_a_f32(x, NFD, row0, 0, lane), frag_b_pk(wp, 4, 0, nt, lane), c);
  int mo = (lane >> 4) << 3;
#pragma unroll
  for (int v = 0; v < 8; ++v) {
    float val = fmaxf(c[v], 0.0f);
    long idx = (long)(row0 + mo + v) * HF + n;
    out[idx] = val;
    h[idx] = val;
  }
}

// ---- hid = relu(edge_attr @ nn_w1 + b1).  M=80000, K=64, N=128 -------------
__global__ void k_hid(const float* __restrict__ ea,
                      const _Float16* __restrict__ w1p,
                      const float* __restrict__ b1, float* __restrict__ hid) {
  int lane = threadIdx.x;
  int mt = blockIdx.x >> 3, nt = blockIdx.x & 7;
  int row0 = mt * 16, col0 = nt * 16;
  int n = col0 + (lane & 15);
  float bv = b1[n];
  v8f c = {bv, bv, bv, bv, bv, bv, bv, bv};
  c = wmma16(frag_a_f32(ea, HF, row0, 0, lane),  frag_b_pk(w1p, 8, 0, nt, lane), c);
  c = wmma16(frag_a_f32(ea, HF, row0, 32, lane), frag_b_pk(w1p, 8, 1, nt, lane), c);
  int mo = (lane >> 4) << 3;
#pragma unroll
  for (int v = 0; v < 8; ++v)
    hid[(long)(row0 + mo + v) * NHID + n] = fmaxf(c[v], 0.0f);
}

// ---- deg[dst] += 1 ---------------------------------------------------------
__global__ void k_deg(const int* __restrict__ ei, float* __restrict__ deg) {
  int e = blockIdx.x * 256 + threadIdx.x;
  if (e < NEDGE) atomicAdd(&deg[ei[NEDGE + e]], 1.0f);
}

// ---- Fused NNConv message: per 16 edges, w_e tile in LDS, then matvec ------
// Phase A: WE[16][4096] = hid_tile @ nn_w2 + b2  (WMMA, f16 in LDS, 128 KB)
// Phase B: msg[e] = out[src[e]] @ reshape(WE[e],64,64); atomicAdd into agg[dst]
__global__ void k_msg(const float* __restrict__ hid,
                      const _Float16* __restrict__ w2p,
                      const float* __restrict__ b2, const float* __restrict__ out,
                      const int* __restrict__ ei, float* __restrict__ agg) {
  extern __shared__ char smem[];
  _Float16* WE = (_Float16*)smem;                               // 16 x 4096 f16
  float* sfeat = (float*)(smem + (size_t)16 * HH * sizeof(_Float16)); // 16 x 64
  int tid = threadIdx.x, lane = tid & 31, wv = tid >> 5;
  int eb = blockIdx.x * 16;

  // gather source node features for these 16 edges
  for (int i = tid; i < 16 * HF; i += 256) {
    int el = i >> 6, cc = i & 63;
    int s = ei[eb + el];
    sfeat[i] = out[(long)s * HF + cc];
  }

  // Phase A: each of 8 waves covers 32 of the 256 N-tiles; K=128 (4 chunks)
  v16h a0 = frag_a_f32(hid, NHID, eb, 0, lane);
  v16h a1 = frag_a_f32(hid, NHID, eb, 32, lane);
  v16h a2 = frag_a_f32(hid, NHID, eb, 64, lane);
  v16h a3 = frag_a_f32(hid, NHID, eb, 96, lane);
  int mo = (lane >> 4) << 3;
  for (int nt = wv; nt < 256; nt += 8) {
    // prefetch next tile's packed-B panel (global_prefetch_b8 on gfx1250)
    if (nt + 8 < 256)
      __builtin_prefetch(w2p + (((size_t)(nt + 8)) * 32 + lane) * 16, 0, 1);
    int n = nt * 16 + (lane & 15);
    float bv = b2[n];
    v8f c = {bv, bv, bv, bv, bv, bv, bv, bv};
    c = wmma16(a0, frag_b_pk(w2p, 256, 0, nt, lane), c);
    c = wmma16(a1, frag_b_pk(w2p, 256, 1, nt, lane), c);
    c = wmma16(a2, frag_b_pk(w2p, 256, 2, nt, lane), c);
    c = wmma16(a3, frag_b_pk(w2p, 256, 3, nt, lane), c);
#pragma unroll
    for (int v = 0; v < 8; ++v) WE[(mo + v) * HH + n] = (_Float16)c[v];
  }
  __syncthreads();

  // Phase B: thread t -> edge t>>4, 4 output cols; msg[e,o] = sum_i s_i WE[e,i*64+o]
  int el = tid >> 4;
  int oc = (tid & 15) << 2;
  float acc0 = 0.f, acc1 = 0.f, acc2 = 0.f, acc3 = 0.f;
  const _Float16* we = WE + el * HH + oc;
  const float* sf = sfeat + el * HF;
#pragma unroll 4
  for (int i = 0; i < HF; ++i) {
    float si = sf[i];
    v4h w4 = *(const v4h*)(we + i * HF);   // 8B-aligned ds_load_b64
    acc0 += si * (float)w4[0];
    acc1 += si * (float)w4[1];
    acc2 += si * (float)w4[2];
    acc3 += si * (float)w4[3];
  }
  int dst = ei[NEDGE + eb + el];
  float* ap = agg + (long)dst * HF + oc;
  atomicAdd(ap + 0, acc0);
  atomicAdd(ap + 1, acc1);
  atomicAdd(ap + 2, acc2);
  atomicAdd(ap + 3, acc3);
}

// ---- Node update: m = relu(agg/deg + out@root + cb); h = GRU(m, h); out = h -
__global__ void k_node(const float* __restrict__ agg, const float* __restrict__ deg,
                       const _Float16* __restrict__ rootp,
                       const float* __restrict__ conv_bias,
                       const _Float16* __restrict__ wihp,
                       const _Float16* __restrict__ whhp,
                       const float* __restrict__ bih, const float* __restrict__ bhh,
                       float* out, float* h) {
  __shared__ float m_lds[16 * HF];     //  4 KB
  __shared__ float gi_lds[16 * 192];   // 12 KB
  __shared__ float gh_lds[16 * 192];   // 12 KB
  int tid = threadIdx.x, lane = tid & 31, wv = tid >> 5;
  int nb = blockIdx.x * 16;
  int mo = (lane >> 4) << 3;

  // Stage 1: waves 0..3 compute the 4 N-tiles of out @ conv_root (K=64)
  if (wv < 4) {
    int col0 = wv * 16;
    int n = col0 + (lane & 15);
    v8f c = {0.f, 0.f, 0.f, 0.f, 0.f, 0.f, 0.f, 0.f};
    c = wmma16(frag_a_f32(out, HF, nb, 0, lane),  frag_b_pk(rootp, 4, 0, wv, lane), c);
    c = wmma16(frag_a_f32(out, HF, nb, 32, lane), frag_b_pk(rootp, 4, 1, wv, lane), c);
    float cb = conv_bias[n];
#pragma unroll
    for (int v = 0; v < 8; ++v) {
      int node = nb + mo + v;
      float inv = 1.0f / fmaxf(deg[node], 1.0f);
      float val = c[v] + cb + agg[(long)node * HF + n] * inv;
      m_lds[(mo + v) * HF + n] = fmaxf(val, 0.0f);
    }
  }
  __syncthreads();

  // Stage 2: 24 tile-jobs (12 for gi = m@wih+bih, 12 for gh = h@whh+bhh)
  for (int j = wv; j < 24; j += 8) {
    bool is_gi = j < 12;
    int nt = is_gi ? j : j - 12;
    int n = nt * 16 + (lane & 15);
    float bv = (is_gi ? bih : bhh)[n];
    v8f c = {bv, bv, bv, bv, bv, bv, bv, bv};
    if (is_gi) {
      c = wmma16(frag_a_f32(m_lds, HF, 0, 0, lane),  frag_b_pk(wihp, 12, 0, nt, lane), c);
      c = wmma16(frag_a_f32(m_lds, HF, 0, 32, lane), frag_b_pk(wihp, 12, 1, nt, lane), c);
    } else {
      c = wmma16(frag_a_f32(h, HF, nb, 0, lane),  frag_b_pk(whhp, 12, 0, nt, lane), c);
      c = wmma16(frag_a_f32(h, HF, nb, 32, lane), frag_b_pk(whhp, 12, 1, nt, lane), c);
    }
    float* dl = is_gi ? gi_lds : gh_lds;
#pragma unroll
    for (int v = 0; v < 8; ++v) dl[(mo + v) * 192 + n] = c[v];
  }
  __syncthreads();

  // Stage 3: GRU elementwise (torch gate order r, z, n)
  for (int i = tid; i < 16 * HF; i += 256) {
    int nl = i >> 6, cc = i & 63;
    float ir = gi_lds[nl * 192 + cc];
    float iz = gi_lds[nl * 192 + 64 + cc];
    float in_ = gi_lds[nl * 192 + 128 + cc];
    float hr = gh_lds[nl * 192 + cc];
    float hz = gh_lds[nl * 192 + 64 + cc];
    float hn = gh_lds[nl * 192 + 128 + cc];
    float r = sigf(ir + hr);
    float z = sigf(iz + hz);
    float nn_ = tanhf(in_ + r * hn);
    long gidx = (long)(nb + nl) * HF + cc;
    float hv = (1.0f - z) * nn_ + z * h[gidx];
    h[gidx] = hv;
    out[gidx] = hv;
  }
}

// ---- Set2Set: LSTM cell, one block per graph (tiny) ------------------------
__global__ void k_lstm(const float* __restrict__ qst, const float* __restrict__ wih,
                       const float* __restrict__ whh, const float* __restrict__ bih,
                       const float* __restrict__ bhh, float* hs, float* cs) {
  __shared__ float g[256];
  int b = blockIdx.x, j = threadIdx.x;  // 256 threads = 4H gates
  float acc = bih[j] + bhh[j];
  for (int k = 0; k < 2 * HF; ++k) acc += qst[b * 128 + k] * wih[k * 256 + j];
  for (int k = 0; k < HF; ++k) acc += hs[b * HF + k] * whh[k * 256 + j];
  g[j] = acc;
  __syncthreads();
  if (j < HF) {  // gate order i, f, g, o
    float ii = g[j], ff = g[64 + j], gg = g[128 + j], oo = g[192 + j];
    float c2 = sigf(ff) * cs[b * HF + j] + sigf(ii) * tanhf(gg);
    float h2 = sigf(oo) * tanhf(c2);
    cs[b * HF + j] = c2;
    hs[b * HF + j] = h2;
  }
}

// ---- Attention logits + per-graph max and exp-sum (block owns graph b) -----
__global__ void k_stats(const float* __restrict__ out, const int* __restrict__ batch,
                        const float* __restrict__ hs, float* __restrict__ en,
                        float* __restrict__ emax, float* __restrict__ esum) {
  __shared__ float red[256];
  __shared__ float q[HF];
  int b = blockIdx.x, tid = threadIdx.x;
  if (tid < HF) q[tid] = hs[b * HF + tid];
  __syncthreads();
  float lmax = -3.4e38f;
  for (int n = tid; n < NNODE; n += 256) {
    if (batch[n] == b) {
      float d = 0.f;
      for (int k = 0; k < HF; ++k) d += out[(long)n * HF + k] * q[k];
      en[n] = d;
      lmax = fmaxf(lmax, d);
    }
  }
  red[tid] = lmax;
  __syncthreads();
  for (int s = 128; s > 0; s >>= 1) {
    if (tid < s) red[tid] = fmaxf(red[tid], red[tid + s]);
    __syncthreads();
  }
  float mx = red[0];
  if (tid == 0) emax[b] = mx;
  __syncthreads();
  float ls = 0.f;
  for (int n = tid; n < NNODE; n += 256)
    if (batch[n] == b) ls += __expf(en[n] - mx);
  red[tid] = ls;
  __syncthreads();
  for (int s = 128; s > 0; s >>= 1) {
    if (tid < s) red[tid] += red[tid + s];
    __syncthreads();
  }
  if (tid == 0) esum[b] = red[0];
}

// ---- r = softmax-weighted sum; q_star = [q, r] -----------------------------
__global__ void k_r(const float* __restrict__ out, const int* __restrict__ batch,
                    const float* __restrict__ en, const float* __restrict__ emax,
                    const float* __restrict__ esum, const float* __restrict__ hs,
                    float* __restrict__ qst) {
  int b = blockIdx.x, t = threadIdx.x;  // 64 threads, one output column each
  float mx = emax[b];
  float inv = 1.0f / (esum[b] + 1e-16f);
  float acc = 0.f;
  for (int n = 0; n < NNODE; ++n) {
    if (batch[n] == b) {
      float a = __expf(en[n] - mx) * inv;
      acc += a * out[(long)n * HF + t];
    }
  }
  qst[b * 128 + t] = hs[b * HF + t];
  qst[b * 128 + 64 + t] = acc;
}

// ---- head: res = relu(q_star@lin1+b1) @ lin2 + b2 --------------------------
__global__ void k_final(const float* __restrict__ qst, const float* __restrict__ w1,
                        const float* __restrict__ b1, const float* __restrict__ w2,
                        const float* __restrict__ b2, float* __restrict__ res) {
  __shared__ float o[HF];
  int b = blockIdx.x, j = threadIdx.x;  // 64 threads
  float acc = b1[j];
  for (int k = 0; k < 2 * HF; ++k) acc += qst[b * 128 + k] * w1[k * HF + j];
  o[j] = fmaxf(acc, 0.0f);
  __syncthreads();
  if (j == 0) {
    float s = b2[0];
    for (int k = 0; k < HF; ++k) s += o[k] * w2[k];
    res[b] = s;
  }
}

extern "C" void kernel_launch(void* const* d_in, const int* in_sizes, int n_in,
                              void* d_out, int out_size, void* d_ws, size_t ws_size,
                              hipStream_t stream) {
  (void)in_sizes; (void)n_in; (void)out_size; (void)ws_size;
  const float* x        = (const float*)d_in[0];
  const int*   ei       = (const int*)d_in[1];
  const int*   batch    = (const int*)d_in[2];
  const float* ea       = (const float*)d_in[3];
  const float* lin0_w   = (const float*)d_in[4];
  const float* lin0_b   = (const float*)d_in[5];
  const float* nn_w1    = (const float*)d_in[6];
  const float* nn_b1    = (const float*)d_in[7];
  const float* nn_w2    = (const float*)d_in[8];
  const float* nn_b2    = (const float*)d_in[9];
  const float* conv_root = (const float*)d_in[10];
  const float* conv_bias = (const float*)d_in[11];
  const float* gru_wih  = (const float*)d_in[12];
  const float* gru_whh  = (const float*)d_in[13];
  const float* gru_bih  = (const float*)d_in[14];
  const float* gru_bhh  = (const float*)d_in[15];
  const float* lstm_wih = (const float*)d_in[16];
  const float* lstm_whh = (const float*)d_in[17];
  const float* lstm_bih = (const float*)d_in[18];
  const float* lstm_bhh = (const float*)d_in[19];
  const float* lin1_w   = (const float*)d_in[20];
  const float* lin1_b   = (const float*)d_in[21];
  const float* lin2_w   = (const float*)d_in[22];
  const float* lin2_b   = (const float*)d_in[23];
  float* res = (float*)d_out;

  float* ws   = (float*)d_ws;
  float* out  = ws; ws += (size_t)NNODE * HF;
  float* h    = ws; ws += (size_t)NNODE * HF;
  float* hid  = ws; ws += (size_t)NEDGE * NHID;
  float* agg  = ws; ws += (size_t)NNODE * HF;
  float* deg  = ws; ws += NNODE;
  float* en   = ws; ws += NNODE;
  float* hs   = ws; ws += NB * HF;      // hs, cs, qst contiguous (one memset)
  float* cs   = ws; ws += NB * HF;
  float* qst  = ws; ws += NB * 2 * HF;
  float* emax = ws; ws += NB;
  float* esum = ws; ws += NB;
  // packed f16 B-operand weights (sizes in halves are multiples of 16 -> each
  // buffer stays 32-byte aligned for v16h loads)
  _Float16* p_w2   = (_Float16*)ws; ws += (NHID * HH) / 2;      // 128x4096
  _Float16* p_w1   = (_Float16*)ws; ws += (HF * NHID) / 2;      // 64x128
  _Float16* p_lin0 = (_Float16*)ws; ws += (NFD * HF) / 2;       // 32x64
  _Float16* p_root = (_Float16*)ws; ws += (HF * HF) / 2;        // 64x64
  _Float16* p_wih  = (_Float16*)ws; ws += (HF * 192) / 2;       // 64x192
  _Float16* p_whh  = (_Float16*)ws; ws += (HF * 192) / 2;       // 64x192

  size_t msg_lds = (size_t)16 * HH * sizeof(_Float16)   // 128 KB WE tile
                 + (size_t)16 * HF * sizeof(float);     // + 4 KB gathered src
  hipFuncSetAttribute((const void*)k_msg,
                      hipFuncAttributeMaxDynamicSharedMemorySize, (int)msg_lds);

  hipMemsetAsync(deg, 0, NNODE * sizeof(float), stream);
  hipMemsetAsync(hs, 0, (size_t)(NB * HF * 2 + NB * 2 * HF) * sizeof(float), stream);

  // one-time weight packing into WMMA B-fragment layout
  k_pack<<<(NHID * HH + 255) / 256, 256, 0, stream>>>(nn_w2, p_w2, NHID, HH);
  k_pack<<<(HF * NHID + 255) / 256, 256, 0, stream>>>(nn_w1, p_w1, HF, NHID);
  k_pack<<<(NFD * HF + 255) / 256, 256, 0, stream>>>(lin0_w, p_lin0, NFD, HF);
  k_pack<<<(HF * HF + 255) / 256, 256, 0, stream>>>(conv_root, p_root, HF, HF);
  k_pack<<<(HF * 192 + 255) / 256, 256, 0, stream>>>(gru_wih, p_wih, HF, 192);
  k_pack<<<(HF * 192 + 255) / 256, 256, 0, stream>>>(gru_whh, p_whh, HF, 192);

  k_deg<<<(NEDGE + 255) / 256, 256, 0, stream>>>(ei, deg);
  k_lin0<<<(NNODE / 16) * 4, 32, 0, stream>>>(x, p_lin0, lin0_b, out, h);
  k_hid<<<(NEDGE / 16) * 8, 32, 0, stream>>>(ea, p_w1, nn_b1, hid);

  for (int it = 0; it < 3; ++it) {
    hipMemsetAsync(agg, 0, (size_t)NNODE * HF * sizeof(float), stream);
    k_msg<<<NEDGE / 16, 256, msg_lds, stream>>>(hid, p_w2, nn_b2, out, ei, agg);
    k_node<<<NNODE / 16, 256, 0, stream>>>(agg, deg, p_root, conv_bias,
                                           p_wih, p_whh, gru_bih, gru_bhh,
                                           out, h);
  }

  for (int it = 0; it < 3; ++it) {
    k_lstm<<<NB, 256, 0, stream>>>(qst, lstm_wih, lstm_whh, lstm_bih, lstm_bhh,
                                   hs, cs);
    k_stats<<<NB, 256, 0, stream>>>(out, batch, hs, en, emax, esum);
    k_r<<<NB, 64, 0, stream>>>(out, batch, en, emax, esum, hs, qst);
  }
  k_final<<<NB, 64, 0, stream>>>(qst, lin1_w, lin1_b, lin2_w, lin2_b, res);
}
